// RPN_14697378087270
// MI455X (gfx1250) — matrix operand order repaired and checked by reading, not compiled, not run on hardware
//
#include <hip/hip_runtime.h>
#include <hip/hip_bf16.h>
#include <math.h>

// ---------------------------------------------------------------------------
// Types for CDNA5 WMMA (wave32): v_wmma_f32_16x16x32_bf16
// ---------------------------------------------------------------------------
typedef __bf16 bf16;
typedef __attribute__((ext_vector_type(16))) __bf16 v16bf;
typedef __attribute__((ext_vector_type(8)))  __bf16 v8bf;
typedef __attribute__((ext_vector_type(8)))  float  v8f;

union AFrag { v16bf v; v8bf h[2]; };

static __device__ __forceinline__ unsigned f2bfbits(float f) {
  union { float f; unsigned u; } x; x.f = f;
  return (x.u + 0x7FFFu + ((x.u >> 16) & 1u)) >> 16;       // RNE
}
static __device__ __forceinline__ bf16 f2bf(float f) {
  union { unsigned short s; bf16 b; } o; o.s = (unsigned short)f2bfbits(f);
  return o.b;
}
static __device__ __forceinline__ unsigned bfpack2(float lo, float hi) {
  return f2bfbits(lo) | (f2bfbits(hi) << 16);
}

// Problem constants
#define NLEV 4
#define NA_TOT 146880
#define PRE_NMS 2000
#define POST_NMS 1000
#define NMS_BLK 63            // ceil(2000/32)

// ---------------------------------------------------------------------------
// Weight packing: conv1 (512x256x3x3) -> bf16 fragments
// Layout: Wp[kt(72)][nt(32)][lane(32)][i(16)], B-frag: n = nt*16 + (lane&15),
// k-in-tile = i + 16*(lane>=16); global k = tap*256 + c (tap-major, c inner).
// ---------------------------------------------------------------------------
__global__ void pack_conv1_w(const float* __restrict__ w, bf16* __restrict__ wp) {
  int t = blockIdx.x * blockDim.x + threadIdx.x;
  if (t >= 72 * 32 * 32 * 16) return;
  int i  = t & 15;
  int ln = (t >> 4) & 31;
  int nt = (t >> 9) & 31;
  int kt = t >> 14;                       // 0..71
  int n  = nt * 16 + (ln & 15);           // 0..511
  int kk = i + ((ln >> 4) << 4);          // 0..31
  int c  = ((kt & 7) << 5) + kk;          // 0..255
  int tap = kt >> 3;                      // 0..8
  int ky = tap / 3, kx = tap % 3;
  wp[t] = f2bf(w[(((size_t)n * 256 + c) * 3 + ky) * 3 + kx]);
}

// Head weights: logits(6) + deltas(24) concatenated, padded to 32 channels.
// Wp[kt(144)][nt(2)][lane(32)][i(16)]
__global__ void pack_head_w(const float* __restrict__ lw, const float* __restrict__ dw,
                            bf16* __restrict__ wp) {
  int t = blockIdx.x * blockDim.x + threadIdx.x;
  if (t >= 144 * 2 * 32 * 16) return;
  int i  = t & 15;
  int ln = (t >> 4) & 31;
  int nt = (t >> 9) & 1;
  int kt = t >> 10;                       // 0..143
  int ch = nt * 16 + (ln & 15);           // 0..31
  int kk = i + ((ln >> 4) << 4);          // 0..31
  int c  = ((kt & 15) << 5) + kk;         // 0..511
  int tap = kt >> 4;                      // 0..8
  int ky = tap / 3, kx = tap % 3;
  float v = 0.f;
  if (ch < 6)       v = lw[(((size_t)ch * 512 + c) * 3 + ky) * 3 + kx];
  else if (ch < 30) v = dw[(((size_t)(ch - 6) * 512 + c) * 3 + ky) * 3 + kx];
  wp[t] = f2bf(v);
}

// ---------------------------------------------------------------------------
// conv1: implicit GEMM, 3x3 SAME, 256->512, ReLU, bf16 out (channel-major).
// Block = 256 threads (8 waves). M-tile = 32 positions; one block covers all
// 512 output channels: wave w owns N-tiles [4w, 4w+3].
// Register blocking: 2 M-subtiles x 4 N-tiles = 8 WMMA per K-step per wave.
// K = 2304 -> 72 K-steps, double-buffered LDS A tile, packed b32 DS stores.
// ---------------------------------------------------------------------------
__global__ void __launch_bounds__(256) conv1_wmma(
    const float* __restrict__ X, const bf16* __restrict__ Wp,
    const float* __restrict__ bias, bf16* __restrict__ Y, int H, int W) {
  const int HW = H * W;
  __shared__ __align__(16) bf16 Alds[2][32][32];   // [buf][m][k] = 4 KB
  const int tid  = threadIdx.x;
  const int wave = tid >> 5, lane = tid & 31;
  const int nt0 = wave * 4;                        // 4 N-tiles per wave
  const int posbase = blockIdx.x * 32;

  auto stage = [&](int kt, int buf) {
    const int tap = kt >> 3;
    const int ky = tap / 3 - 1, kx = tap % 3 - 1;
    const int c0 = (kt & 7) << 5;
#pragma unroll
    for (int e = 0; e < 2; ++e) {
      int t = (tid << 1) + e;          // 0..511 (pairs)
      int m  = t >> 4;                 // position 0..31
      int kp = (t & 15) << 1;          // k pair base 0,2,..30
      int pos = posbase + m;
      int y = pos / W + ky;
      int x = pos % W + kx;
      unsigned pk = 0u;
      if ((unsigned)y < (unsigned)H && (unsigned)x < (unsigned)W) {
        size_t base = (size_t)(c0 + kp) * HW + (size_t)y * W + x;
        pk = bfpack2(X[base], X[base + HW]);
      }
      *(unsigned*)&Alds[buf][m][kp] = pk;
    }
  };

  v8f acc[2][4];
#pragma unroll
  for (int ms = 0; ms < 2; ++ms)
#pragma unroll
    for (int nn = 0; nn < 4; ++nn)
      acc[ms][nn] = (v8f){0.f,0.f,0.f,0.f,0.f,0.f,0.f,0.f};

  stage(0, 0);
  __syncthreads();

  const int ml   = lane & 15;
  const int koff = (lane >> 4) << 4;   // 0 or 16
  for (int kt = 0; kt < 72; ++kt) {
    AFrag a[2];
#pragma unroll
    for (int ms = 0; ms < 2; ++ms) {
      a[ms].h[0] = *(const v8bf*)&Alds[kt & 1][ms * 16 + ml][koff];
      a[ms].h[1] = *(const v8bf*)&Alds[kt & 1][ms * 16 + ml][koff + 8];
    }
    v16bf b[4];
#pragma unroll
    for (int nn = 0; nn < 4; ++nn)
      b[nn] = *(const v16bf*)(Wp + (((size_t)(kt * 32 + nt0 + nn) * 32 + lane) << 4));
    if (kt + 1 < 72) {
      __builtin_prefetch(Wp + (((size_t)((kt + 1) * 32 + nt0) * 32 + lane) << 4), 0, 1);
      stage(kt + 1, (kt + 1) & 1);
    }
#pragma unroll
    for (int ms = 0; ms < 2; ++ms)
#pragma unroll
      for (int nn = 0; nn < 4; ++nn)
        acc[ms][nn] = __builtin_amdgcn_wmma_f32_16x16x32_bf16(
            false, a[ms].v, false, b[nn], (short)0, acc[ms][nn], false, false);
    __syncthreads();
  }

  // C layout: lanes 0-15: M=r, N=lane; lanes 16-31: M=8+r, N=lane-16.
  const int mbase = (lane >> 4) << 3;
#pragma unroll
  for (int nn = 0; nn < 4; ++nn) {
    const int n = (nt0 + nn) * 16 + ml;
    const float bn = bias[n];
#pragma unroll
    for (int ms = 0; ms < 2; ++ms) {
      const int pos0 = posbase + ms * 16 + mbase;
#pragma unroll
      for (int r = 0; r < 8; ++r) {
        float v = acc[ms][nn][r] + bn;
        v = v > 0.f ? v : 0.f;
        Y[(size_t)n * HW + pos0 + r] = f2bf(v);
      }
    }
  }
}

// ---------------------------------------------------------------------------
// Head conv: 512 -> 32 (6 logits + 24 deltas + pad), raw f32 out [HW][32].
// Block = 128 threads (4 waves): wave = M-subtile, each wave owns BOTH
// N-tiles -> 2 WMMA per K-step. M-tile = 64. K = 4608 -> 144 K-steps.
// ---------------------------------------------------------------------------
__global__ void __launch_bounds__(128) conv_head_wmma(
    const bf16* __restrict__ F, const bf16* __restrict__ Wp,
    const float* __restrict__ lb, const float* __restrict__ db,
    float* __restrict__ Out, int H, int W) {
  const int HW = H * W;
  __shared__ __align__(16) bf16 Alds[2][64][32];   // 8 KB
  const int tid  = threadIdx.x;
  const int wave = tid >> 5, lane = tid & 31;      // wave = msub 0..3
  const int posbase = blockIdx.x * 64;

  auto stage = [&](int kt, int buf) {
    const int tap = kt >> 4;
    const int ky = tap / 3 - 1, kx = tap % 3 - 1;
    const int c0 = (kt & 15) << 5;
#pragma unroll
    for (int e = 0; e < 8; ++e) {
      int t = (tid << 3) + e;          // 0..1023 (pairs)
      int m  = t >> 4;                 // 0..63
      int kp = (t & 15) << 1;          // 0,2,..30
      int pos = posbase + m;
      int y = pos / W + ky;
      int x = pos % W + kx;
      unsigned pk = 0u;
      if ((unsigned)y < (unsigned)H && (unsigned)x < (unsigned)W) {
        size_t base = (size_t)(c0 + kp) * HW + (size_t)y * W + x;
        union { bf16 b[2]; unsigned u; } p;
        p.b[0] = F[base];
        p.b[1] = F[base + HW];
        pk = p.u;
      }
      *(unsigned*)&Alds[buf][m][kp] = pk;
    }
  };

  v8f acc[2];
  acc[0] = (v8f){0.f,0.f,0.f,0.f,0.f,0.f,0.f,0.f};
  acc[1] = (v8f){0.f,0.f,0.f,0.f,0.f,0.f,0.f,0.f};
  stage(0, 0);
  __syncthreads();

  const int mloc = wave * 16 + (lane & 15);
  const int koff = (lane >> 4) << 4;
  for (int kt = 0; kt < 144; ++kt) {
    AFrag a;
    a.h[0] = *(const v8bf*)&Alds[kt & 1][mloc][koff];
    a.h[1] = *(const v8bf*)&Alds[kt & 1][mloc][koff + 8];
    v16bf b0 = *(const v16bf*)(Wp + (((size_t)(kt * 2 + 0) * 32 + lane) << 4));
    v16bf b1 = *(const v16bf*)(Wp + (((size_t)(kt * 2 + 1) * 32 + lane) << 4));
    if (kt + 1 < 144) {
      __builtin_prefetch(Wp + (((size_t)((kt + 1) * 2) * 32 + lane) << 4), 0, 1);
      stage(kt + 1, (kt + 1) & 1);
    }
    acc[0] = __builtin_amdgcn_wmma_f32_16x16x32_bf16(false, a.v, false, b0,
                                                     (short)0, acc[0], false, false);
    acc[1] = __builtin_amdgcn_wmma_f32_16x16x32_bf16(false, a.v, false, b1,
                                                     (short)0, acc[1], false, false);
    __syncthreads();
  }

  const int mbase = wave * 16 + ((lane >> 4) << 3);
#pragma unroll
  for (int nt = 0; nt < 2; ++nt) {
    const int ch = nt * 16 + (lane & 15);
    float bn = 0.f;
    if (ch < 6) bn = lb[ch]; else if (ch < 30) bn = db[ch - 6];
#pragma unroll
    for (int r = 0; r < 8; ++r)
      Out[(size_t)(posbase + mbase + r) * 32 + ch] = acc[nt][r] + bn;
  }
}

// ---------------------------------------------------------------------------
// Softmax fg score + fg delta extraction per position.
// Head channels: [0..5] logits (pairs per base), [6..29] deltas (b*8+cls*4+j).
// ---------------------------------------------------------------------------
__global__ void score_kernel(const float* __restrict__ raw, float* __restrict__ scores,
                             float* __restrict__ fgd, int HW, int aoff) {
  int q = blockIdx.x * blockDim.x + threadIdx.x;
  if (q >= HW) return;
  const float* r = raw + (size_t)q * 32;
#pragma unroll
  for (int b = 0; b < 3; ++b) {
    float l0 = r[2 * b], l1 = r[2 * b + 1];
    float p = 1.f / (1.f + __expf(l0 - l1));      // softmax fg
    int a = aoff + q * 3 + b;
    scores[a] = p;
#pragma unroll
    for (int j = 0; j < 4; ++j)
      fgd[(size_t)a * 4 + j] = r[6 + b * 8 + 4 + j];   // cls==1 deltas
  }
}

// ---------------------------------------------------------------------------
// Iterative top-2000 (single block, LDS argmax reduction). Matches jax
// top_k ordering (descending score, lower index wins ties). Mutates scores.
// ---------------------------------------------------------------------------
__global__ void __launch_bounds__(1024) topk_kernel(float* __restrict__ scores,
                                                    int* __restrict__ topi,
                                                    float* __restrict__ tops) {
  __shared__ float sv[1024];
  __shared__ int   si[1024];
  const int tid = threadIdx.x;
  for (int it = 0; it < PRE_NMS; ++it) {
    float best = -1e30f; int bi = 0x7fffffff;
    for (int a = tid; a < NA_TOT; a += 1024) {
      float s = scores[a];
      if (s > best) { best = s; bi = a; }
    }
    sv[tid] = best; si[tid] = bi;
    __syncthreads();
    for (int off = 512; off > 0; off >>= 1) {
      if (tid < off) {
        float so = sv[tid + off]; int io = si[tid + off];
        if (so > sv[tid] || (so == sv[tid] && io < si[tid])) { sv[tid] = so; si[tid] = io; }
      }
      __syncthreads();
    }
    if (tid == 0) {
      topi[it] = si[0];
      tops[it] = sv[0];
      scores[si[0]] = -1e30f;
    }
    __syncthreads();
  }
}

// ---------------------------------------------------------------------------
// Anchor generation + delta decode + clip for the top-2000.
// ---------------------------------------------------------------------------
__global__ void decode_kernel(const int* __restrict__ topi, const float* __restrict__ fgd,
                              float* __restrict__ boxes) {
  int t = blockIdx.x * blockDim.x + threadIdx.x;
  if (t >= PRE_NMS) return;
  const int offs[5] = {0, 110592, 138240, 145152, 146880};
  const int strd[4] = {4, 8, 16, 32};
  const int Ws[4]   = {192, 96, 48, 24};
  int a = topi[t];
  int l = 0;
  while (l < 3 && a >= offs[l + 1]) ++l;
  int q = a - offs[l];
  int b = q % 3;
  int cell = q / 3;
  int W = Ws[l];
  int y = cell / W, x = cell % W;
  float s = (float)strd[l];
  float size = s * 8.f;
  float ar = (b == 0) ? 0.5f : ((b == 1) ? 1.f : 2.f);
  float sq = __fsqrt_rn(ar);
  float hh = size * sq, ww = size / sq;
  float cx = ((float)x + 0.5f) * s;
  float cy = ((float)y + 0.5f) * s;
  float dx = fgd[(size_t)a * 4 + 0], dy = fgd[(size_t)a * 4 + 1];
  float dw = fgd[(size_t)a * 4 + 2], dh = fgd[(size_t)a * 4 + 3];
  float ncx = dx * ww + cx;
  float ncy = dy * hh + cy;
  float nw = ww * __expf(fminf(fmaxf(dw, -4.f), 4.f));
  float nh = hh * __expf(fminf(fmaxf(dh, -4.f), 4.f));
  const float XM = 767.f, YM = 767.f;
  boxes[t * 4 + 0] = fminf(fmaxf(ncx - 0.5f * nw, 0.f), XM);
  boxes[t * 4 + 1] = fminf(fmaxf(ncy - 0.5f * nh, 0.f), YM);
  boxes[t * 4 + 2] = fminf(fmaxf(ncx + 0.5f * nw, 0.f), XM);
  boxes[t * 4 + 3] = fminf(fmaxf(ncy + 0.5f * nh, 0.f), YM);
}

// ---------------------------------------------------------------------------
// NMS suppression bitmask: one wave32 per (rowBlock, colBlock).
// ---------------------------------------------------------------------------
__global__ void nms_mask_kernel(const float* __restrict__ boxes, unsigned* __restrict__ mask) {
  __shared__ float cbx[32][4];
  const int rb = blockIdx.x, cblk = blockIdx.y;
  const int lane = threadIdx.x;
  int col = cblk * 32 + lane;
  if (col < PRE_NMS) {
#pragma unroll
    for (int j = 0; j < 4; ++j) cbx[lane][j] = boxes[col * 4 + j];
  } else {
#pragma unroll
    for (int j = 0; j < 4; ++j) cbx[lane][j] = 0.f;
  }
  __syncthreads();
  int row = rb * 32 + lane;
  unsigned bits = 0;
  if (row < PRE_NMS) {
    float rx0 = boxes[row * 4 + 0], ry0 = boxes[row * 4 + 1];
    float rx1 = boxes[row * 4 + 2], ry1 = boxes[row * 4 + 3];
    float ra = (rx1 - rx0) * (ry1 - ry0);
    for (int j = 0; j < 32; ++j) {
      int c = cblk * 32 + j;
      if (c <= row || c >= PRE_NMS) continue;
      float iw = fminf(rx1, cbx[j][2]) - fmaxf(rx0, cbx[j][0]);
      float ih = fminf(ry1, cbx[j][3]) - fmaxf(ry0, cbx[j][1]);
      iw = fmaxf(iw, 0.f); ih = fmaxf(ih, 0.f);
      float inter = iw * ih;
      float ca = (cbx[j][2] - cbx[j][0]) * (cbx[j][3] - cbx[j][1]);
      float iou = inter / (ra + ca - inter + 1e-8f);
      if (iou > 0.7f) bits |= (1u << j);
    }
  }
  if (row < PRE_NMS) mask[(size_t)row * NMS_BLK + cblk] = bits;
}

// ---------------------------------------------------------------------------
// Sequential NMS reduce + final selection, mirroring
//   s = where(keep, top_s, -1); top_k(s, 1000)
// Output rows: [img_idx(0), x0, y0, x1, y1, score].
// ---------------------------------------------------------------------------
__global__ void nms_finalize(const unsigned* __restrict__ mask, const float* __restrict__ boxes,
                             const float* __restrict__ tops, float* __restrict__ out) {
  if (threadIdx.x != 0 || blockIdx.x != 0) return;
  unsigned remv[NMS_BLK], kb[NMS_BLK];
  for (int i = 0; i < NMS_BLK; ++i) { remv[i] = 0u; kb[i] = 0u; }
  int outp = 0;
  for (int i = 0; i < PRE_NMS; ++i) {
    if (!((remv[i >> 5] >> (i & 31)) & 1u)) {
      kb[i >> 5] |= (1u << (i & 31));
      if (outp < POST_NMS) {
        float* o = out + (size_t)outp * 6;
        o[0] = 0.f;
        o[1] = boxes[i * 4 + 0]; o[2] = boxes[i * 4 + 1];
        o[3] = boxes[i * 4 + 2]; o[4] = boxes[i * 4 + 3];
        o[5] = tops[i];
        ++outp;
      }
      for (int j = 0; j < NMS_BLK; ++j) remv[j] |= mask[(size_t)i * NMS_BLK + j];
    }
  }
  for (int i = 0; i < PRE_NMS && outp < POST_NMS; ++i) {
    if (!((kb[i >> 5] >> (i & 31)) & 1u)) {
      float* o = out + (size_t)outp * 6;
      o[0] = 0.f;
      o[1] = boxes[i * 4 + 0]; o[2] = boxes[i * 4 + 1];
      o[3] = boxes[i * 4 + 2]; o[4] = boxes[i * 4 + 3];
      o[5] = -1.f;
      ++outp;
    }
  }
  while (outp < POST_NMS) {
    float* o = out + (size_t)outp * 6;
    for (int j = 0; j < 5; ++j) o[j] = 0.f;
    o[5] = -1.f;
    ++outp;
  }
}

// ---------------------------------------------------------------------------
// Host-side orchestration
// ---------------------------------------------------------------------------
extern "C" void kernel_launch(void* const* d_in, const int* in_sizes, int n_in,
                              void* d_out, int out_size, void* d_ws, size_t ws_size,
                              hipStream_t stream) {
  (void)in_sizes; (void)n_in; (void)out_size; (void)ws_size;
  const float* feats[NLEV] = {(const float*)d_in[1], (const float*)d_in[2],
                              (const float*)d_in[3], (const float*)d_in[4]};
  const float* conv_w  = (const float*)d_in[5];
  const float* conv_b  = (const float*)d_in[6];
  const float* logit_w = (const float*)d_in[7];
  const float* logit_b = (const float*)d_in[8];
  const float* delta_w = (const float*)d_in[9];
  const float* delta_b = (const float*)d_in[10];
  float* out = (float*)d_out;

  const int Hs[NLEV]   = {192, 96, 48, 24};
  const int HWs[NLEV]  = {36864, 9216, 2304, 576};
  const int aoff[NLEV] = {0, 110592, 138240, 145152};

  char* ws = (char*)d_ws;
  size_t off = 0;
  auto take = [&](size_t bytes) -> void* {
    void* p = ws + off;
    off = (off + bytes + 255) & ~(size_t)255;
    return p;
  };

  bf16* wp1[NLEV];  for (int l = 0; l < NLEV; ++l) wp1[l]  = (bf16*)take((size_t)72 * 32 * 32 * 16 * 2);
  bf16* wp2[NLEV];  for (int l = 0; l < NLEV; ++l) wp2[l]  = (bf16*)take((size_t)144 * 2 * 32 * 16 * 2);
  bf16* fbuf[NLEV]; for (int l = 0; l < NLEV; ++l) fbuf[l] = (bf16*)take((size_t)512 * HWs[l] * 2);
  float* hraw[NLEV];for (int l = 0; l < NLEV; ++l) hraw[l] = (float*)take((size_t)HWs[l] * 32 * 4);
  float* scores = (float*)take((size_t)NA_TOT * 4);
  float* fgd    = (float*)take((size_t)NA_TOT * 4 * 4);
  int*   topi   = (int*)take((size_t)PRE_NMS * 4);
  float* tops   = (float*)take((size_t)PRE_NMS * 4);
  float* boxes  = (float*)take((size_t)PRE_NMS * 4 * 4);
  unsigned* msk = (unsigned*)take((size_t)PRE_NMS * NMS_BLK * 4);

  // 1) pack weights
  for (int l = 0; l < NLEV; ++l) {
    pack_conv1_w<<<4608, 256, 0, stream>>>(conv_w + (size_t)l * 512 * 256 * 9, wp1[l]);
    pack_head_w<<<576, 256, 0, stream>>>(logit_w + (size_t)l * 6 * 512 * 9,
                                         delta_w + (size_t)l * 24 * 512 * 9, wp2[l]);
  }
  // 2) conv1 + ReLU (bf16 WMMA implicit GEMM, 8 WMMA/K-step per wave)
  for (int l = 0; l < NLEV; ++l) {
    conv1_wmma<<<HWs[l] / 32, 256, 0, stream>>>(feats[l], wp1[l],
                                                conv_b + (size_t)l * 512,
                                                fbuf[l], Hs[l], Hs[l]);
  }
  // 3) fused logit+delta head conv (2 WMMA/K-step per wave)
  for (int l = 0; l < NLEV; ++l) {
    conv_head_wmma<<<HWs[l] / 64, 128, 0, stream>>>(fbuf[l], wp2[l],
                                                    logit_b + (size_t)l * 6,
                                                    delta_b + (size_t)l * 24,
                                                    hraw[l], Hs[l], Hs[l]);
  }
  // 4) softmax fg scores + fg deltas
  for (int l = 0; l < NLEV; ++l) {
    score_kernel<<<(HWs[l] + 255) / 256, 256, 0, stream>>>(hraw[l], scores, fgd,
                                                           HWs[l], aoff[l]);
  }
  // 5) top-2000
  topk_kernel<<<1, 1024, 0, stream>>>(scores, topi, tops);
  // 6) decode + clip
  decode_kernel<<<(PRE_NMS + 255) / 256, 256, 0, stream>>>(topi, fgd, boxes);
  // 7) NMS bitmask + sequential reduce + final top-1000 layout
  dim3 mg(NMS_BLK, NMS_BLK);
  nms_mask_kernel<<<mg, 32, 0, stream>>>(boxes, msk);
  nms_finalize<<<1, 32, 0, stream>>>(msk, boxes, tops, out);
}
